// ODE2VAE_6717328851341
// MI455X (gfx1250) — matrix operand order repaired and checked by reading, not compiled, not run on hardware
//
#include <hip/hip_runtime.h>
#include <hip/hip_bf16.h>
#include <math.h>

// ---------------------------------------------------------------------------
// ODE2VAE forward for MI455X (gfx1250, wave32, WMMA).
// Heavy matmuls (decoder transposed convs as implicit GEMM, FC layers) use
// v_wmma_f32_16x16x32_f16 fed from pre-packed fp16 operands. Deconv weight
// tiles are staged into LDS with the CDNA5 async global->LDS path
// (global_load_async_to_lds_b128 + s_wait_asynccnt).
// ---------------------------------------------------------------------------

typedef __attribute__((ext_vector_type(16))) _Float16 v16h;
typedef __attribute__((ext_vector_type(8)))  _Float16 v8h;
typedef __attribute__((ext_vector_type(8)))  float    v8f;
typedef __attribute__((address_space(3)))    _Float16 lds_f16;

#define LOG2PI_F 1.8378770664093453f
#define QDIM 8
#define HD 50
#define DTSTEP 0.1f

// ---------------------------------------------------------------------------
__global__ void o2v_zero_acc(float* acc) {
  if (threadIdx.x < 8) acc[threadIdx.x] = 0.0f;
}

// ---------------------------------------------------------------------------
// Direct conv (encoder only: tiny channel counts).  act: 0 none, 1 relu
// ---------------------------------------------------------------------------
__global__ void o2v_conv2d(const float* __restrict__ x, const float* __restrict__ w,
                           const float* __restrict__ b, float* __restrict__ y,
                           int NB, int Cin, int Hin, int Win,
                           int Cout, int Hout, int Wout,
                           int ksz, int stride, int pad,
                           long long inSampleStride, int act) {
  size_t idx = (size_t)blockIdx.x * blockDim.x + threadIdx.x;
  size_t total = (size_t)NB * Cout * Hout * Wout;
  if (idx >= total) return;
  int wo = (int)(idx % Wout); size_t t = idx / Wout;
  int ho = (int)(t % Hout); t /= Hout;
  int co = (int)(t % Cout); int nb = (int)(t / Cout);
  float s = b[co];
  for (int ci = 0; ci < Cin; ++ci) {
    for (int kh = 0; kh < ksz; ++kh) {
      int hi = ho * stride - pad + kh;
      if (hi < 0 || hi >= Hin) continue;
      for (int kw = 0; kw < ksz; ++kw) {
        int wi = wo * stride - pad + kw;
        if (wi < 0 || wi >= Win) continue;
        float xv = x[(size_t)nb * inSampleStride + ((size_t)ci * Hin + hi) * Win + wi];
        float wv = w[(((size_t)co * Cin + ci) * ksz + kh) * ksz + kw];
        s = fmaf(xv, wv, s);
      }
    }
  }
  if (act == 1) s = fmaxf(s, 0.0f);
  y[idx] = s;
}

// ---------------------------------------------------------------------------
// Training-mode BatchNorm stats (per-channel mean / biased var), no divides
// in the hot loop.
// ---------------------------------------------------------------------------
__global__ void o2v_bn_stats(const float* __restrict__ x, float* __restrict__ mean,
                             float* __restrict__ var, int NB, int C, int HW) {
  __shared__ float rs[256], rs2[256];
  int c = blockIdx.x;
  float s = 0.0f, s2 = 0.0f;
  for (int nb = 0; nb < NB; ++nb) {
    const float* p = x + ((size_t)nb * C + c) * HW;
    for (int i = threadIdx.x; i < HW; i += blockDim.x) {
      float v = p[i];
      s += v; s2 += v * v;
    }
  }
  rs[threadIdx.x] = s; rs2[threadIdx.x] = s2;
  __syncthreads();
  for (int off = 128; off > 0; off >>= 1) {
    if ((int)threadIdx.x < off) { rs[threadIdx.x] += rs[threadIdx.x + off]; rs2[threadIdx.x] += rs2[threadIdx.x + off]; }
    __syncthreads();
  }
  if (threadIdx.x == 0) {
    float cnt = (float)NB * (float)HW;
    float m = rs[0] / cnt;
    mean[c] = m;
    var[c] = rs2[0] / cnt - m * m;
  }
}

// BN + ReLU in place; optionally also writes an fp16 mirror for the next
// WMMA consumer.
__global__ void o2v_bn_apply_relu(float* __restrict__ x, const float* __restrict__ mean,
                                  const float* __restrict__ var, const float* __restrict__ g,
                                  const float* __restrict__ b, _Float16* __restrict__ mir,
                                  size_t total, int C, int HW) {
  size_t idx = (size_t)blockIdx.x * blockDim.x + threadIdx.x;
  if (idx >= total) return;
  int c = (int)((idx / HW) % C);
  float v = x[idx];
  float y = fmaxf(g[c] * (v - mean[c]) * rsqrtf(var[c] + 1e-5f) + b[c], 0.0f);
  x[idx] = y;
  if (mir) mir[idx] = (_Float16)y;
}

// ---------------------------------------------------------------------------
// Operand packing kernels (fp32 -> fp16, zero-padded K to a multiple of 32).
// ---------------------------------------------------------------------------
__global__ void o2v_pack_a(const float* __restrict__ A, _Float16* __restrict__ out,
                           int M, int K, int Kpad) {
  size_t idx = (size_t)blockIdx.x * blockDim.x + threadIdx.x;
  size_t total = (size_t)M * Kpad;
  if (idx >= total) return;
  int k = (int)(idx % Kpad); size_t m = idx / Kpad;
  out[idx] = (k < K) ? (_Float16)A[m * K + k] : (_Float16)0.0f;
}

// B[K,N] fp32 -> Bt[N,Kpad] fp16 (transposed so fragment loads are contiguous)
__global__ void o2v_pack_bt(const float* __restrict__ B, _Float16* __restrict__ out,
                            int K, int N, int Kpad) {
  size_t idx = (size_t)blockIdx.x * blockDim.x + threadIdx.x;
  size_t total = (size_t)N * Kpad;
  if (idx >= total) return;
  int k = (int)(idx % Kpad); size_t n = idx / Kpad;
  out[idx] = (k < K) ? (_Float16)B[(size_t)k * N + n] : (_Float16)0.0f;
}

// ConvTranspose weight w[Cin,Cout,k,k] fp32 -> Wt[CoutPad, Kpad] fp16,
// K index = ci*k*k + kh*k + kw.
__global__ void o2v_pack_wt(const float* __restrict__ w, _Float16* __restrict__ out,
                            int Cin, int Cout, int ksz, int CoutPad, int Kpad) {
  size_t idx = (size_t)blockIdx.x * blockDim.x + threadIdx.x;
  size_t total = (size_t)CoutPad * Kpad;
  if (idx >= total) return;
  int kk = (int)(idx % Kpad); int co = (int)(idx / Kpad);
  int Ktot = Cin * ksz * ksz;
  float v = 0.0f;
  if (kk < Ktot && co < Cout) {
    int kw = kk % ksz; int t = kk / ksz;
    int kh = t % ksz; int ci = t / ksz;
    v = w[(((size_t)ci * Cout + co) * ksz + kh) * ksz + kw];
  }
  out[idx] = (_Float16)v;
}

// ---------------------------------------------------------------------------
// WMMA GEMM on packed fp16 operands: C[M,N] = Ah[M,Kpad] * Bt[N,Kpad]^T + bias
// 128 threads = 4 waves, one 16x16 C tile per wave.
// Fragment layouts (CDNA5 ISA 7.12.2):
//   A f16 16x32: lane L row=L%16; halves = K{(L<16?0:16)+0..15} (contiguous, as B; not the ISA table)
//   B f16 32x16: lane L col=L%16; halves = K{(L<16?0:16)+0..15} (this code's order, not the ISA table)
//   C f32 16x16: vgpr r, lane L -> row=r+8*(L>>4), col=L%16
// act: 0 none, 1 relu.  Ch (optional) = fp16 mirror of C.
// ---------------------------------------------------------------------------
__global__ void o2v_gemm_wmma(const _Float16* __restrict__ Ah, const _Float16* __restrict__ Bt,
                              const float* __restrict__ bias, float* __restrict__ C,
                              _Float16* __restrict__ Ch,
                              int M, int N, int Kpad, int act) {
  const int wave = threadIdx.x >> 5;
  const int tileM = (blockIdx.x * 4 + wave) * 16;
  const int tileN = blockIdx.y * 16;
  const int lane = threadIdx.x & 31;
  const int lidx = lane & 15;
  const int hi = lane >> 4;

  const _Float16* arow = Ah + (size_t)(tileM + lidx) * Kpad + (hi << 4);
  const _Float16* brow = Bt + (size_t)(tileN + lidx) * Kpad + (hi << 4);

  v8f acc = {0.f, 0.f, 0.f, 0.f, 0.f, 0.f, 0.f, 0.f};
#pragma unroll 2
  for (int k0 = 0; k0 < Kpad; k0 += 32) {
    __builtin_prefetch(arow + k0 + 64, 0, 1);
    v8h a0 = *(const v8h*)(arow + k0);
    v8h a1 = *(const v8h*)(arow + k0 + 8);
    v8h b0 = *(const v8h*)(brow + k0);
    v8h b1 = *(const v8h*)(brow + k0 + 8);
    v16h af = __builtin_shufflevector(a0, a1, 0, 1, 2, 3, 4, 5, 6, 7, 8, 9, 10, 11, 12, 13, 14, 15);
    v16h bf = __builtin_shufflevector(b0, b1, 0, 1, 2, 3, 4, 5, 6, 7, 8, 9, 10, 11, 12, 13, 14, 15);
    acc = __builtin_amdgcn_wmma_f32_16x16x32_f16(false, af, false, bf, (short)0, acc, false, false);
  }
  const int col = tileN + lidx;
  const float bv = bias[col];
#pragma unroll
  for (int r = 0; r < 8; ++r) {
    int row = tileM + r + (hi << 3);
    float v = acc[r] + bv;
    if (act == 1) v = fmaxf(v, 0.0f);
    C[(size_t)row * N + col] = v;
    if (Ch) Ch[(size_t)row * N + col] = (_Float16)v;
  }
}

// ---------------------------------------------------------------------------
// WMMA implicit-GEMM transposed conv on fp16 activations + packed weights.
// y[n,co,ho,wo] = sum_{ci,kh,kw} x[n,ci,(ho+p-kh)/s,(wo+p-kw)/s] * w[ci,co,kh,kw]
// The 16 weight rows for this block's tileN are staged into LDS once per
// block via async global->LDS DMA (ASYNCcnt), then B fragments come from
// LDS (ds_load_b128). A gather is branchless (clamped index + cndmask).
// 128 threads = 4 waves, one M tile per wave.  act: 0 none, 2 sigmoid.
// ---------------------------------------------------------------------------
__global__ void o2v_deconv_wmma(const _Float16* __restrict__ xh, const _Float16* __restrict__ wt,
                                const float* __restrict__ bias, float* __restrict__ y,
                                int NB, int Cin, int Hin, int Win,
                                int Cout, int Hout, int Wout,
                                int ksz, int sshift, int pad,
                                int Kpad, int act) {
  __shared__ __align__(16) _Float16 wlds[16 * 1600];   // 51.2 KB weight tile
  __shared__ int kdec[1664];                           // per-K tap decode
  const int tileN = blockIdx.y * 16;

  // ---- async-stage the weight tile: wt[tileN .. tileN+15][0..Kpad) -> LDS ----
  {
    lds_f16* ldst = (lds_f16*)wlds;
    const _Float16* gsrc = wt + (size_t)tileN * Kpad;
    const int chunks = (16 * Kpad) >> 3;               // 16-byte chunks
    for (int c = threadIdx.x; c < chunks; c += blockDim.x) {
      asm volatile("global_load_async_to_lds_b128 %0, %1, off"
                   :: "v"(ldst + (c << 3)), "v"(gsrc + (c << 3)) : "memory");
    }
  }
  // ---- fill the tap-decode table while the DMA is in flight ----
  const int Ktot = Cin * ksz * ksz;
  for (int i = threadIdx.x; i < Ktot; i += blockDim.x) {
    int kw = i % ksz; int t = i / ksz;
    kdec[i] = ((t / ksz) << 16) | ((t % ksz) << 8) | kw;
  }
  asm volatile("s_wait_asynccnt 0x0" ::: "memory");
  __syncthreads();

  const int wave = threadIdx.x >> 5;
  const int tileM = (blockIdx.x * 4 + wave) * 16;
  const int lane = threadIdx.x & 31;
  const int lidx = lane & 15;
  const int hi = lane >> 4;
  const int smask = (1 << sshift) - 1;

  // output pixel for this lane's A-fragment row
  int m = tileM + lidx;
  int wo = m % Wout; int tq = m / Wout;
  int ho = tq % Hout; int nb = tq / Hout;
  const int hopp = ho + pad, wopp = wo + pad;
  const size_t xbase = (size_t)nb * Cin * Hin * Win;
  const int co = tileN + lidx;
  const _Float16* brow = &wlds[(size_t)lidx * Kpad + (hi << 4)];  // LDS

  v8f acc = {0.f, 0.f, 0.f, 0.f, 0.f, 0.f, 0.f, 0.f};
#pragma unroll 2
  for (int k0 = 0; k0 < Kpad; k0 += 32) {
    v16h af;
#pragma unroll
    for (int hh = 0; hh < 16; ++hh) {
      int kk = k0 + (hi << 4) + hh;
      int inb = (kk < Ktot) ? 1 : 0;
      int dec = kdec[inb ? kk : 0];
      int kw = dec & 255, kh = (dec >> 8) & 255, ci = dec >> 16;
      int hq = hopp - kh, wq = wopp - kw;
      int hin = hq >> sshift, win = wq >> sshift;
      bool ok = inb && (hq >= 0) && (wq >= 0) &&
                ((hq & smask) == 0) && ((wq & smask) == 0) &&
                (hin < Hin) && (win < Win);
      size_t idx = ok ? (xbase + ((size_t)ci * Hin + hin) * Win + win) : xbase;
      _Float16 hv = xh[idx];                       // unconditional load
      af[hh] = ok ? hv : (_Float16)0.0f;           // cndmask, no branch
    }
    v8h b0 = *(const v8h*)(brow + k0);             // ds_load_b128
    v8h b1 = *(const v8h*)(brow + k0 + 8);
    v16h bf = __builtin_shufflevector(b0, b1, 0, 1, 2, 3, 4, 5, 6, 7, 8, 9, 10, 11, 12, 13, 14, 15);
    acc = __builtin_amdgcn_wmma_f32_16x16x32_f16(false, af, false, bf, (short)0, acc, false, false);
  }
  if (co < Cout) {
    const float bv = bias[co];
#pragma unroll
    for (int r = 0; r < 8; ++r) {
      int row = tileM + r + (hi << 3);
      int wo2 = row % Wout; int t2 = row / Wout;
      int ho2 = t2 % Hout; int nb2 = t2 / Hout;
      float v = acc[r] + bv;
      if (act == 2) v = 1.0f / (1.0f + expf(-v));
      y[(((size_t)nb2 * Cout + co) * Hout + ho2) * Wout + wo2] = v;
    }
  }
}

// ---------------------------------------------------------------------------
// z0 = qz0_m + eps * exp(qz0_logv);  logp0 = -0.5*|eps|^2 - Q*log(2pi)
// ---------------------------------------------------------------------------
__global__ void o2v_z0(const float* __restrict__ qm, const float* __restrict__ qlv,
                       const float* __restrict__ eps, float* __restrict__ z0,
                       float* __restrict__ lp0, int N) {
  int n = blockIdx.x * blockDim.x + threadIdx.x;
  if (n >= N) return;
  float se = 0.0f;
  for (int j = 0; j < 2 * QDIM; ++j) {
    float e = eps[n * 2 * QDIM + j];
    z0[n * 2 * QDIM + j] = qm[n * 2 * QDIM + j] + e * expf(qlv[n * 2 * QDIM + j]);
    se += e * e;
  }
  lp0[n] = -0.5f * se - (float)QDIM * LOG2PI_F;
}

// ---------------------------------------------------------------------------
// Euler integration of the BNN ODE with forward-mode Jacobian trace.
// One wave per block, one thread per sample; per-thread arrays in LDS
// laid out [j][32] (bank-conflict-free). Weights reparameterized into LDS.
// ---------------------------------------------------------------------------
#define PT(A, j) A[(j) * 32 + tid]

__global__ void o2v_ode(const float* __restrict__ z0in, const float* __restrict__ lp0in,
                        const float* Wm0, const float* Wls0, const float* bm0, const float* bls0,
                        const float* Wm1, const float* Wls1, const float* bm1, const float* bls1,
                        const float* Wm2, const float* Wls2, const float* bm2, const float* bls2,
                        const float* eW0, const float* eb0, const float* eW1,
                        const float* eb1, const float* eW2, const float* eb2,
                        float* __restrict__ zt_out, float* __restrict__ st_out,
                        float* __restrict__ klz_acc, int N, int T) {
  __shared__ float sW0[16 * HD], sB0[HD], sW1[HD * HD], sB1[HD], sW2[HD * QDIM], sB2[QDIM];
  __shared__ float sc1[HD * 32], sk1[HD * 32], sh1[HD * 32];
  __shared__ float sc2[HD * 32], sk2[HD * 32];
  __shared__ float st1[HD * 32], st2[HD * 32];

  const int tid = threadIdx.x;
  for (int i = tid; i < 16 * HD; i += 32) sW0[i] = Wm0[i] + expf(Wls0[i]) * eW0[i];
  for (int i = tid; i < HD; i += 32)      sB0[i] = bm0[i] + expf(bls0[i]) * eb0[i];
  for (int i = tid; i < HD * HD; i += 32) sW1[i] = Wm1[i] + expf(Wls1[i]) * eW1[i];
  for (int i = tid; i < HD; i += 32)      sB1[i] = bm1[i] + expf(bls1[i]) * eb1[i];
  for (int i = tid; i < HD * QDIM; i += 32) sW2[i] = Wm2[i] + expf(Wls2[i]) * eW2[i];
  for (int i = tid; i < QDIM; i += 32)    sB2[i] = bm2[i] + expf(bls2[i]) * eb2[i];
  __syncthreads();

  const int n = blockIdx.x * 32 + tid;
  if (n >= N) return;
  float z[2 * QDIM];
  for (int j = 0; j < 2 * QDIM; ++j) z[j] = z0in[n * 2 * QDIM + j];
  float lp = lp0in[n];
  float klz = 0.0f;

  for (int t = 0; t < T; ++t) {
    // scan collects carry BEFORE the update
    float se = 0.0f;
    for (int j = 0; j < 2 * QDIM; ++j) {
      zt_out[((size_t)n * T + t) * 2 * QDIM + j] = z[j];
      se += z[j] * z[j];
    }
    for (int j = 0; j < QDIM; ++j)
      st_out[((size_t)n * T + t) * QDIM + j] = z[QDIM + j];
    float lpz = -0.5f * se - (float)QDIM * LOG2PI_F;
    klz += (lp - lpz);

    // layer 1
    float mean1 = 0.0f;
    for (int j = 0; j < HD; ++j) {
      float a = sB0[j];
      for (int i = 0; i < 2 * QDIM; ++i) a = fmaf(z[i], sW0[i * HD + j], a);
      PT(sc1, j) = a; mean1 += a;
    }
    mean1 *= (1.0f / HD);
    float var1 = 0.0f;
    for (int j = 0; j < HD; ++j) { float c = PT(sc1, j) - mean1; PT(sc1, j) = c; var1 += c * c; }
    var1 *= (1.0f / HD);
    const float invs1 = rsqrtf(var1 + 1e-5f);
    for (int j = 0; j < HD; ++j) {
      float xx = PT(sc1, j) * invs1;
      float kk = (xx > 0.0f) ? 1.0f : expf(xx);
      PT(sk1, j) = kk;
      PT(sh1, j) = (xx > 0.0f) ? xx : (kk - 1.0f);
    }

    // layer 2 + head
    float f[QDIM];
    for (int o = 0; o < QDIM; ++o) f[o] = sB2[o];
    float mean2 = 0.0f;
    for (int j = 0; j < HD; ++j) {
      float a = sB1[j];
      for (int i = 0; i < HD; ++i) a = fmaf(PT(sh1, i), sW1[i * HD + j], a);
      PT(sc2, j) = a; mean2 += a;
    }
    mean2 *= (1.0f / HD);
    float var2 = 0.0f;
    for (int j = 0; j < HD; ++j) { float c = PT(sc2, j) - mean2; PT(sc2, j) = c; var2 += c * c; }
    var2 *= (1.0f / HD);
    const float invs2 = rsqrtf(var2 + 1e-5f);
    for (int j = 0; j < HD; ++j) {
      float xx = PT(sc2, j) * invs2;
      float kk = (xx > 0.0f) ? 1.0f : expf(xx);
      PT(sk2, j) = kk;
      float h2 = (xx > 0.0f) ? xx : (kk - 1.0f);
      for (int o = 0; o < QDIM; ++o) f[o] = fmaf(h2, sW2[j * QDIM + o], f[o]);
    }

    // trace(d f / d v) via 8 forward-mode tangents
    const float i31 = invs1 * invs1 * invs1;
    const float i32v = invs2 * invs2 * invs2;
    float tr = 0.0f;
    for (int d = 0; d < QDIM; ++d) {
      float mt = 0.0f;
      for (int j = 0; j < HD; ++j) { float tv = sW0[d * HD + j]; PT(st1, j) = tv; mt += tv; }
      mt *= (1.0f / HD);
      float q = 0.0f;
      for (int j = 0; j < HD; ++j) q += PT(sc1, j) * PT(st1, j);
      q *= (1.0f / HD);
      for (int j = 0; j < HD; ++j)
        PT(st1, j) = PT(sk1, j) * ((PT(st1, j) - mt) * invs1 - PT(sc1, j) * q * i31);
      float mt2 = 0.0f;
      for (int j = 0; j < HD; ++j) {
        float a = 0.0f;
        for (int u = 0; u < HD; ++u) a = fmaf(PT(st1, u), sW1[u * HD + j], a);
        PT(st2, j) = a; mt2 += a;
      }
      mt2 *= (1.0f / HD);
      float q2 = 0.0f;
      for (int j = 0; j < HD; ++j) q2 += PT(sc2, j) * PT(st2, j);
      q2 *= (1.0f / HD);
      float tf = 0.0f;
      for (int j = 0; j < HD; ++j)
        tf += PT(sk2, j) * ((PT(st2, j) - mt2) * invs2 - PT(sc2, j) * q2 * i32v) * sW2[j * QDIM + d];
      tr += tf;
    }

    // Euler update: s += DT*v (old v) first, then v += DT*f
    for (int j = 0; j < QDIM; ++j) z[QDIM + j] += DTSTEP * z[j];
    for (int j = 0; j < QDIM; ++j) z[j] += DTSTEP * f[j];
    lp -= DTSTEP * tr;
  }
  atomicAdd(klz_acc, klz);
}

// ---------------------------------------------------------------------------
__global__ void o2v_lhood(const float* __restrict__ Xrec, const float* __restrict__ X,
                          float* __restrict__ acc, long long total) {
  __shared__ float red[256];
  float s = 0.0f;
  for (long long i = (long long)blockIdx.x * blockDim.x + threadIdx.x; i < total;
       i += (long long)gridDim.x * blockDim.x) {
    float xr = Xrec[i], xv = X[i];
    s += logf(0.001f + xr) * xv + logf(1.001f - xr) * (1.0f - xv);
  }
  red[threadIdx.x] = s; __syncthreads();
  for (int off = 128; off > 0; off >>= 1) {
    if ((int)threadIdx.x < off) red[threadIdx.x] += red[threadIdx.x + off];
    __syncthreads();
  }
  if (threadIdx.x == 0) atomicAdd(acc, red[0]);
}

__global__ void o2v_klw(const float* __restrict__ m, const float* __restrict__ ls,
                        int n, float* __restrict__ acc) {
  __shared__ float red[256];
  float s = 0.0f;
  for (int i = threadIdx.x; i < n; i += blockDim.x) {
    float lsv = ls[i], mv = m[i];
    s += 0.5f * (expf(2.0f * lsv) + mv * mv - 1.0f) - lsv;
  }
  red[threadIdx.x] = s; __syncthreads();
  for (int off = 128; off > 0; off >>= 1) {
    if ((int)threadIdx.x < off) red[threadIdx.x] += red[threadIdx.x + off];
    __syncthreads();
  }
  if (threadIdx.x == 0) atomicAdd(acc, red[0]);
}

__global__ void o2v_finalize(const float* __restrict__ acc, const int* __restrict__ Ndata,
                             float* __restrict__ out, int N) {
  float nd = (float)Ndata[0];
  float lh = nd * acc[0] / (float)N;
  float kz = nd * acc[1] / (float)N;
  float kw = acc[2];
  out[0] = lh - kz - kw;  // elbo
  out[1] = lh;            // lhood_t
  out[2] = kz;            // kl_z_t
  out[3] = kw;            // kl_w
}

// ---------------------------------------------------------------------------
// Host launcher
// ---------------------------------------------------------------------------
static inline unsigned cdiv256(size_t n) { return (unsigned)((n + 255) / 256); }

extern "C" void kernel_launch(void* const* d_in, const int* in_sizes, int n_in,
                              void* d_out, int out_size, void* d_ws, size_t ws_size,
                              hipStream_t stream) {
  (void)in_sizes; (void)n_in; (void)out_size; (void)ws_size;

  // ---- inputs (setup_inputs dict order; nested dicts in insertion order) ----
  const float* X      = (const float*)d_in[0];   // [512,16,1,28,28]
  const int*   Ndata  = (const int*)d_in[1];
  const float* eps_z  = (const float*)d_in[3];   // [512,16]
  const float* c1_w = (const float*)d_in[4];  const float* c1_b = (const float*)d_in[5];
  const float* bn1g = (const float*)d_in[6];  const float* bn1b = (const float*)d_in[7];
  const float* c2_w = (const float*)d_in[8];  const float* c2_b = (const float*)d_in[9];
  const float* bn2g = (const float*)d_in[10]; const float* bn2b = (const float*)d_in[11];
  const float* c3_w = (const float*)d_in[12]; const float* c3_b = (const float*)d_in[13];
  const float* fc1w = (const float*)d_in[14]; const float* fc1b = (const float*)d_in[15];
  const float* fc2w = (const float*)d_in[16]; const float* fc2b = (const float*)d_in[17];
  const float* fc3w = (const float*)d_in[18]; const float* fc3b = (const float*)d_in[19];
  const float* Wm0 = (const float*)d_in[20]; const float* Wls0 = (const float*)d_in[21];
  const float* bm0 = (const float*)d_in[22]; const float* bls0 = (const float*)d_in[23];
  const float* Wm1 = (const float*)d_in[24]; const float* Wls1 = (const float*)d_in[25];
  const float* bm1 = (const float*)d_in[26]; const float* bls1 = (const float*)d_in[27];
  const float* Wm2 = (const float*)d_in[28]; const float* Wls2 = (const float*)d_in[29];
  const float* bm2 = (const float*)d_in[30]; const float* bls2 = (const float*)d_in[31];
  const float* d1w = (const float*)d_in[32]; const float* d1b = (const float*)d_in[33];
  const float* db1g = (const float*)d_in[34]; const float* db1b = (const float*)d_in[35];
  const float* d2w = (const float*)d_in[36]; const float* d2b = (const float*)d_in[37];
  const float* db2g = (const float*)d_in[38]; const float* db2b = (const float*)d_in[39];
  const float* d3w = (const float*)d_in[40]; const float* d3b = (const float*)d_in[41];
  const float* db3g = (const float*)d_in[42]; const float* db3b = (const float*)d_in[43];
  const float* d4w = (const float*)d_in[44]; const float* d4b = (const float*)d_in[45];
  const float* eW0 = (const float*)d_in[46]; const float* eb0 = (const float*)d_in[47];
  const float* eW1 = (const float*)d_in[48]; const float* eb1 = (const float*)d_in[49];
  const float* eW2 = (const float*)d_in[50]; const float* eb2 = (const float*)d_in[51];

  const int N = 512, T = 16, MNT = 512 * 16;   // 8192 decoder samples

  // ---- output layout (floats, return-order concat) ----
  float* out   = (float*)d_out;
  float* Xrec  = out;                               // 6,422,528
  float* qm    = out + (size_t)6422528;             // [512,16]
  float* qlv   = qm + 8192;                         // [512,16]
  float* ztL   = qlv + 8192;                        // [1,512,16,16]
  float* scal  = ztL + 131072;                      // elbo, lhood_t, kl_z_t, kl_w

  // ---- fp32 workspace ----
  float* ws = (float*)d_ws;
  size_t off = 0;
  float* acc    = ws + off; off += 8;               // [0]=lhood [1]=klz [2]=klw
  float* bnm    = ws + off; off += 64;
  float* bnv    = ws + off; off += 64;
  float* enc1   = ws + off; off += (size_t)512 * 8 * 14 * 14;
  float* enc2   = ws + off; off += (size_t)512 * 16 * 7 * 7;
  float* hbuf   = ws + off; off += (size_t)512 * 512;
  float* z0buf  = ws + off; off += (size_t)512 * 16;
  float* lp0buf = ws + off; off += 512;
  float* stbuf  = ws + off; off += (size_t)MNT * 8;
  float* sbuf   = ws + off; off += (size_t)MNT * 512;
  float* dec1   = ws + off; off += (size_t)MNT * 64 * 36;     // [8192,64,6,6]
  float* dec2   = ws + off; off += (size_t)MNT * 32 * 169;    // [8192,32,13,13]
  float* dec3   = ws + off; off += (size_t)MNT * 16 * 784;    // [8192,16,28,28]

  // ---- fp16 workspace (packed operands + activation mirrors) ----
  _Float16* hws = (_Float16*)(ws + off);
  size_t ho = 0;
  _Float16* hbuf_h = hws + ho; ho += (size_t)512 * 512;        // A of fc1/fc2
  _Float16* fc1wt  = hws + ho; ho += (size_t)16 * 512;         // Bt [16,512]
  _Float16* fc2wt  = hws + ho; ho += (size_t)16 * 512;
  _Float16* st_h   = hws + ho; ho += (size_t)MNT * 32;         // A of fc3 (K 8->32)
  _Float16* fc3wt  = hws + ho; ho += (size_t)512 * 32;         // Bt [512,32]
  _Float16* s_h    = hws + ho; ho += (size_t)MNT * 512;        // d1 input mirror
  _Float16* d1wt   = hws + ho; ho += (size_t)64 * 288;
  _Float16* d2wt   = hws + ho; ho += (size_t)32 * 1600;
  _Float16* d3wt   = hws + ho; ho += (size_t)16 * 800;
  _Float16* d4wt   = hws + ho; ho += (size_t)16 * 416;
  _Float16* dec1_h = hws + ho; ho += (size_t)MNT * 64 * 36;
  _Float16* dec2_h = hws + ho; ho += (size_t)MNT * 32 * 169;
  _Float16* dec3_h = hws + ho; ho += (size_t)MNT * 16 * 784;

  o2v_zero_acc<<<1, 32, 0, stream>>>(acc);

  // ---- weight packing (independent of activations) ----
  o2v_pack_bt<<<cdiv256((size_t)16 * 512), 256, 0, stream>>>(fc1w, fc1wt, 512, 16, 512);
  o2v_pack_bt<<<cdiv256((size_t)16 * 512), 256, 0, stream>>>(fc2w, fc2wt, 512, 16, 512);
  o2v_pack_bt<<<cdiv256((size_t)512 * 32), 256, 0, stream>>>(fc3w, fc3wt, 8, 512, 32);
  o2v_pack_wt<<<cdiv256((size_t)64 * 288), 256, 0, stream>>>(d1w, d1wt, 32, 64, 3, 64, 288);
  o2v_pack_wt<<<cdiv256((size_t)32 * 1600), 256, 0, stream>>>(d2w, d2wt, 64, 32, 5, 32, 1600);
  o2v_pack_wt<<<cdiv256((size_t)16 * 800), 256, 0, stream>>>(d3w, d3wt, 32, 16, 5, 16, 800);
  o2v_pack_wt<<<cdiv256((size_t)16 * 416), 256, 0, stream>>>(d4w, d4wt, 16, 1, 5, 16, 416);

  // ---------------- encoder on X[:,0] ----------------
  {
    size_t tot = (size_t)512 * 8 * 14 * 14;
    o2v_conv2d<<<cdiv256(tot), 256, 0, stream>>>(
        X, c1_w, c1_b, enc1, 512, 1, 28, 28, 8, 14, 14, 5, 2, 2, (long long)T * 784, 0);
    o2v_bn_stats<<<8, 256, 0, stream>>>(enc1, bnm, bnv, 512, 8, 196);
    o2v_bn_apply_relu<<<cdiv256(tot), 256, 0, stream>>>(enc1, bnm, bnv, bn1g, bn1b,
                                                        (_Float16*)nullptr, tot, 8, 196);
  }
  {
    size_t tot = (size_t)512 * 16 * 7 * 7;
    o2v_conv2d<<<cdiv256(tot), 256, 0, stream>>>(
        enc1, c2_w, c2_b, enc2, 512, 8, 14, 14, 16, 7, 7, 5, 2, 2, 8LL * 196, 0);
    o2v_bn_stats<<<16, 256, 0, stream>>>(enc2, bnm, bnv, 512, 16, 49);
    o2v_bn_apply_relu<<<cdiv256(tot), 256, 0, stream>>>(enc2, bnm, bnv, bn2g, bn2b,
                                                        (_Float16*)nullptr, tot, 16, 49);
  }
  {
    size_t tot = (size_t)512 * 32 * 4 * 4;
    o2v_conv2d<<<cdiv256(tot), 256, 0, stream>>>(
        enc2, c3_w, c3_b, hbuf, 512, 16, 7, 7, 32, 4, 4, 5, 2, 2, 16LL * 49, 1);
    o2v_pack_a<<<cdiv256((size_t)512 * 512), 256, 0, stream>>>(hbuf, hbuf_h, 512, 512, 512);
  }

  // ---------------- q(z0) heads + reparameterization ----------------
  o2v_gemm_wmma<<<dim3(512 / 64, 1), 128, 0, stream>>>(hbuf_h, fc1wt, fc1b, qm,
                                                       (_Float16*)nullptr, 512, 16, 512, 0);
  o2v_gemm_wmma<<<dim3(512 / 64, 1), 128, 0, stream>>>(hbuf_h, fc2wt, fc2b, qlv,
                                                       (_Float16*)nullptr, 512, 16, 512, 0);
  o2v_z0<<<dim3((N + 63) / 64), 64, 0, stream>>>(qm, qlv, eps_z, z0buf, lp0buf, N);

  // ---------------- ODE integration ----------------
  o2v_ode<<<dim3(N / 32), 32, 0, stream>>>(
      z0buf, lp0buf,
      Wm0, Wls0, bm0, bls0, Wm1, Wls1, bm1, bls1, Wm2, Wls2, bm2, bls2,
      eW0, eb0, eW1, eb1, eW2, eb2,
      ztL, stbuf, acc + 1, N, T);
  o2v_pack_a<<<cdiv256((size_t)MNT * 32), 256, 0, stream>>>(stbuf, st_h, MNT, 8, 32);

  // ---------------- decoder ----------------
  // fc3: [8192,8(->32)] @ [8,512]; writes fp32 sbuf + fp16 mirror s_h
  o2v_gemm_wmma<<<dim3(MNT / 64, 512 / 16), 128, 0, stream>>>(st_h, fc3wt, fc3b, sbuf,
                                                              s_h, MNT, 512, 32, 0);
  // d1: 32->64, k3 s1 p0 : [8192,32,4,4] -> [8192,64,6,6]
  o2v_deconv_wmma<<<dim3((unsigned)((size_t)MNT * 36 / 64), 4), 128, 0, stream>>>(
      s_h, d1wt, d1b, dec1, MNT, 32, 4, 4, 64, 6, 6, 3, 0, 0, 288, 0);
  {
    size_t tot = (size_t)MNT * 64 * 36;
    o2v_bn_stats<<<64, 256, 0, stream>>>(dec1, bnm, bnv, MNT, 64, 36);
    o2v_bn_apply_relu<<<cdiv256(tot), 256, 0, stream>>>(dec1, bnm, bnv, db1g, db1b,
                                                        dec1_h, tot, 64, 36);
  }
  // d2: 64->32, k5 s2 p1 : -> [8192,32,13,13]
  o2v_deconv_wmma<<<dim3((unsigned)((size_t)MNT * 169 / 64), 2), 128, 0, stream>>>(
      dec1_h, d2wt, d2b, dec2, MNT, 64, 6, 6, 32, 13, 13, 5, 1, 1, 1600, 0);
  {
    size_t tot = (size_t)MNT * 32 * 169;
    o2v_bn_stats<<<32, 256, 0, stream>>>(dec2, bnm, bnv, MNT, 32, 169);
    o2v_bn_apply_relu<<<cdiv256(tot), 256, 0, stream>>>(dec2, bnm, bnv, db2g, db2b,
                                                        dec2_h, tot, 32, 169);
  }
  // d3: 32->16, k5 s2 p1 (op1) : -> [8192,16,28,28]
  o2v_deconv_wmma<<<dim3((unsigned)((size_t)MNT * 784 / 64), 1), 128, 0, stream>>>(
      dec2_h, d3wt, d3b, dec3, MNT, 32, 13, 13, 16, 28, 28, 5, 1, 1, 800, 0);
  {
    size_t tot = (size_t)MNT * 16 * 784;
    o2v_bn_stats<<<16, 256, 0, stream>>>(dec3, bnm, bnv, MNT, 16, 784);
    o2v_bn_apply_relu<<<cdiv256(tot), 256, 0, stream>>>(dec3, bnm, bnv, db3g, db3b,
                                                        dec3_h, tot, 16, 784);
  }
  // d4: 16->1, k5 s1 p2, sigmoid -> Xrec
  o2v_deconv_wmma<<<dim3((unsigned)((size_t)MNT * 784 / 64), 1), 128, 0, stream>>>(
      dec3_h, d4wt, d4b, Xrec, MNT, 16, 28, 28, 1, 28, 28, 5, 0, 2, 416, 2);

  // ---------------- ELBO terms ----------------
  o2v_lhood<<<4096, 256, 0, stream>>>(Xrec, X, acc + 0, 6422528LL);
  o2v_klw<<<1, 256, 0, stream>>>(Wm0, Wls0, 16 * 50, acc + 2);
  o2v_klw<<<1, 256, 0, stream>>>(bm0, bls0, 50, acc + 2);
  o2v_klw<<<1, 256, 0, stream>>>(Wm1, Wls1, 50 * 50, acc + 2);
  o2v_klw<<<1, 256, 0, stream>>>(bm1, bls1, 50, acc + 2);
  o2v_klw<<<1, 256, 0, stream>>>(Wm2, Wls2, 50 * 8, acc + 2);
  o2v_klw<<<1, 256, 0, stream>>>(bm2, bls2, 8, acc + 2);
  o2v_finalize<<<1, 1, 0, stream>>>(acc, Ndata, scal, N);
}
